// GNN_86509231276701
// MI455X (gfx1250) — matrix operand order repaired and checked by reading, not compiled, not run on hardware
//
#include <hip/hip_runtime.h>
#include <hip/hip_bf16.h>

#define NN 50000
#define NE 800000
#define NG 512
#define FIN 128
#define HD 64

typedef __attribute__((ext_vector_type(16))) __bf16        v16bf;
typedef __attribute__((ext_vector_type(8)))  unsigned int  v8u;
typedef __attribute__((ext_vector_type(8)))  float         v8f;

__device__ __forceinline__ unsigned short f2bf(float f) {
  unsigned u = __builtin_bit_cast(unsigned, f);
  u += 0x7fffu + ((u >> 16) & 1u);          // round-to-nearest-even to bf16
  return (unsigned short)(u >> 16);
}
__device__ __forceinline__ unsigned packbf(float lo, float hi) {
  return (unsigned)f2bf(lo) | ((unsigned)f2bf(hi) << 16);
}

// ---------------- utility kernels ----------------
__global__ void k_zero(float* __restrict__ p, int n) {
  int i = blockIdx.x * blockDim.x + threadIdx.x;
  if (i < n) p[i] = 0.0f;
}

__global__ void k_degree(const int* __restrict__ edges, float* __restrict__ deg,
                         int E, int M) {
  int m = blockIdx.x * blockDim.x + threadIdx.x;
  if (m >= M) return;
  int d = (m < E) ? edges[E + m] : (m - E);   // self-loop edges appended
  atomicAdd(&deg[d], 1.0f);
}

__global__ void k_dinv(const float* __restrict__ deg, float* __restrict__ dinv, int n) {
  int i = blockIdx.x * blockDim.x + threadIdx.x;
  if (i < n) dinv[i] = rsqrtf(fmaxf(deg[i], 1.0f));
}

// ---------------- WMMA GEMM: Y[n,HOUT] = X[n,K] @ W[K,HOUT] (bf16 MACs, f32 acc)
// block = 128 threads = 4 waves; each block does 64 rows, each wave 16 rows.
// Compile-time K/HOUT -> fully unrolled, accumulators stay in VGPRs.
// LDS holds bf16 pairs packed in dwords:
//   sXp[r][kk]  (kk = k/2), row stride AS = K/2+4  (16B-aligned, bank-conflict-free)
//   sWp[c][kk]  column-major, col stride WS = K/2+4
template <int K, int HOUT>
__global__ __launch_bounds__(128) void k_gemm_wmma(
    const float* __restrict__ X, const float* __restrict__ Wm,
    float* __restrict__ Y, int nrows) {
  constexpr int K2 = K / 2;
  constexpr int AS = K2 + 4;
  constexpr int WS = K2 + 4;
  __shared__ __align__(16) unsigned sXp[64 * AS];
  __shared__ __align__(16) unsigned sWp[HOUT * WS];

  const int tid = threadIdx.x;
  const int r0  = blockIdx.x * 64;

  // speculative prefetch of the next row-tile (global_prefetch_b8 path)
  __builtin_prefetch(X + (size_t)(r0 + 64) * K, 0, 0);

  // stage W: coalesced read of two K-rows, packed column-major into LDS
  for (int i = tid; i < HOUT * K2; i += 128) {
    int c  = i % HOUT;
    int kk = i / HOUT;
    float lo = Wm[(size_t)(2 * kk) * HOUT + c];
    float hi = Wm[(size_t)(2 * kk + 1) * HOUT + c];
    sWp[c * WS + kk] = packbf(lo, hi);
  }
  // stage X tile: coalesced float2 reads, packed row-major
  for (int i = tid; i < 64 * K2; i += 128) {
    int r  = i / K2;
    int kk = i - r * K2;
    int gr = r0 + r;
    float lo = 0.0f, hi = 0.0f;
    if (gr < nrows) {
      const float2 v = *(const float2*)(X + (size_t)gr * K + 2 * kk);
      lo = v.x; hi = v.y;
    }
    sXp[r * AS + kk] = packbf(lo, hi);
  }
  __syncthreads();

  const int lane = tid & 31;
  const int wv   = tid >> 5;
  const int m    = lane & 15;     // row (A) / col (B,C)
  const int half = lane >> 4;     // K-half select (A,B) / M+8 select (C)
  const int rloc = wv * 16 + m;   // local A row
  constexpr int NCT = HOUT / 16;
  constexpr int NKC = K / 32;

  v8f acc[NCT];
#pragma unroll
  for (int ct = 0; ct < NCT; ++ct) acc[ct] = {};

#pragma unroll
  for (int kc = 0; kc < NKC; ++kc) {
    // A fragment: 8 contiguous dwords -> two ds_load_b128
    const uint4* pa = (const uint4*)&sXp[rloc * AS + kc * 16 + half * 8];
    uint4 a0 = pa[0];
    uint4 a1 = pa[1];
    v8u a32 = {a0.x, a0.y, a0.z, a0.w, a1.x, a1.y, a1.z, a1.w};
    v16bf a = __builtin_bit_cast(v16bf, a32);

#pragma unroll
    for (int ct = 0; ct < NCT; ++ct) {
      // B fragment: 8 contiguous dwords -> two ds_load_b128
      int c = ct * 16 + m;
      const uint4* pb = (const uint4*)&sWp[c * WS + kc * 16 + half * 8];
      uint4 b0 = pb[0];
      uint4 b1 = pb[1];
      v8u b32 = {b0.x, b0.y, b0.z, b0.w, b1.x, b1.y, b1.z, b1.w};
      v16bf b = __builtin_bit_cast(v16bf, b32);
      acc[ct] = __builtin_amdgcn_wmma_f32_16x16x32_bf16(
          false, a, false, b, (short)0, acc[ct], false, false);
    }
  }

  // C/D layout: VGPR v -> M = 8*half + v ; lane&15 -> N
#pragma unroll
  for (int ct = 0; ct < NCT; ++ct) {
#pragma unroll
    for (int v = 0; v < 8; ++v) {
      int row = r0 + wv * 16 + half * 8 + v;
      int col = ct * 16 + m;
      if (row < nrows) Y[(size_t)row * HOUT + col] = acc[ct][v];
    }
  }
}

// ---------------- edge scatter: agg[dst] += xw[src] * dinv[src]*dinv[dst]
// 16 threads per edge, float4 per thread (64 features).
__global__ void k_scatter(const float* __restrict__ xw, const int* __restrict__ edges,
                          const float* __restrict__ dinv, float* __restrict__ agg,
                          int E, int M) {
  long long g = (long long)blockIdx.x * blockDim.x + threadIdx.x;
  int e = (int)(g >> 4);
  if (e >= M) return;
  int s, d;
  if (e < E) { s = edges[e]; d = edges[E + e]; }
  else       { s = e - E;   d = s; }
  float w = dinv[s] * dinv[d];
  int f = ((int)g & 15) * 4;
  float4 v = *(const float4*)(xw + (size_t)s * HD + f);
  float* q = agg + (size_t)d * HD + f;
  atomicAdd(q + 0, v.x * w);
  atomicAdd(q + 1, v.y * w);
  atomicAdd(q + 2, v.z * w);
  atomicAdd(q + 3, v.w * w);
}

__global__ void k_bias_relu(float* __restrict__ A, const float* __restrict__ b,
                            int n, int fmask) {
  int i = blockIdx.x * blockDim.x + threadIdx.x;
  if (i >= n) return;
  A[i] = fmaxf(A[i] + b[i & fmask], 0.0f);
}

__global__ void k_pool(const float* __restrict__ h, const int* __restrict__ batch,
                       float* __restrict__ psum, int n) {
  long long g = (long long)blockIdx.x * blockDim.x + threadIdx.x;
  int node = (int)(g >> 4);
  if (node >= n) return;
  int f  = ((int)g & 15) * 4;
  int bg = batch[node];
  float4 v = *(const float4*)(h + (size_t)node * HD + f);
  float* q = psum + (size_t)bg * HD + f;
  atomicAdd(q + 0, v.x);
  atomicAdd(q + 1, v.y);
  atomicAdd(q + 2, v.z);
  atomicAdd(q + 3, v.w);
}

__global__ void k_count(const int* __restrict__ batch, float* __restrict__ pcnt, int n) {
  int i = blockIdx.x * blockDim.x + threadIdx.x;
  if (i < n) atomicAdd(&pcnt[batch[i]], 1.0f);
}

__global__ void k_divide(float* __restrict__ psum, const float* __restrict__ pcnt, int n) {
  int i = blockIdx.x * blockDim.x + threadIdx.x;
  if (i < n) psum[i] /= fmaxf(pcnt[i >> 6], 1.0f);
}

__global__ void k_final(const float* __restrict__ A, const float* __restrict__ w,
                        const float* __restrict__ b, float* __restrict__ out, int G) {
  int g = blockIdx.x * blockDim.x + threadIdx.x;
  if (g >= G) return;
  float s = b[0];
#pragma unroll 4
  for (int k = 0; k < 128; ++k) s += A[(size_t)g * 128 + k] * w[k];
  out[g] = s;
}

// ---------------- launcher ----------------
static inline int cdiv(long long a, long long b) { return (int)((a + b - 1) / b); }

extern "C" void kernel_launch(void* const* d_in, const int* in_sizes, int n_in,
                              void* d_out, int out_size, void* d_ws, size_t ws_size,
                              hipStream_t stream) {
  const float* x    = (const float*)d_in[0];
  const int*  edges = (const int*)d_in[1];   // [2, NE] : row0 = src, row1 = dst
  const int*  batch = (const int*)d_in[2];
  const float* W1  = (const float*)d_in[3];
  const float* b1  = (const float*)d_in[4];
  const float* W2  = (const float*)d_in[5];
  const float* b2  = (const float*)d_in[6];
  const float* Wf1 = (const float*)d_in[7];
  const float* bf1 = (const float*)d_in[8];
  const float* Wf2 = (const float*)d_in[9];
  const float* bf2 = (const float*)d_in[10];
  float* out = (float*)d_out;

  float* ws   = (float*)d_ws;
  float* deg  = ws;                                  // NN
  float* dinv = deg  + NN;                           // NN
  float* bufA = dinv + NN;                           // NN*HD
  float* bufB = bufA + (size_t)NN * HD;              // NN*HD
  float* psum = bufB + (size_t)NN * HD;              // NG*HD
  float* pcnt = psum + NG * HD;                      // NG
  float* mlp1 = pcnt + NG;                           // NG*128

  const int M  = NE + NN;                            // edges + self loops
  const int NH = NN * HD;

  // normalization
  k_zero  <<<cdiv(NN, 256), 256, 0, stream>>>(deg, NN);
  k_degree<<<cdiv(M, 256),  256, 0, stream>>>(edges, deg, NE, M);
  k_dinv  <<<cdiv(NN, 256), 256, 0, stream>>>(deg, dinv, NN);

  // layer 1: xw = x @ W1 ; agg = scatter ; h1 = relu(agg + b1)
  k_gemm_wmma<FIN, HD><<<cdiv(NN, 64), 128, 0, stream>>>(x, W1, bufA, NN);
  k_zero     <<<cdiv(NH, 256), 256, 0, stream>>>(bufB, NH);
  k_scatter  <<<cdiv((long long)M * 16, 256), 256, 0, stream>>>(bufA, edges, dinv, bufB, NE, M);
  k_bias_relu<<<cdiv(NH, 256), 256, 0, stream>>>(bufB, b1, NH, HD - 1);

  // layer 2
  k_gemm_wmma<HD, HD><<<cdiv(NN, 64), 128, 0, stream>>>(bufB, W2, bufA, NN);
  k_zero     <<<cdiv(NH, 256), 256, 0, stream>>>(bufB, NH);
  k_scatter  <<<cdiv((long long)M * 16, 256), 256, 0, stream>>>(bufA, edges, dinv, bufB, NE, M);
  k_bias_relu<<<cdiv(NH, 256), 256, 0, stream>>>(bufB, b2, NH, HD - 1);

  // global mean pool
  k_zero  <<<cdiv(NG * HD + NG, 256), 256, 0, stream>>>(psum, NG * HD + NG); // psum + pcnt
  k_pool  <<<cdiv((long long)NN * 16, 256), 256, 0, stream>>>(bufB, batch, psum, NN);
  k_count <<<cdiv(NN, 256), 256, 0, stream>>>(batch, pcnt, NN);
  k_divide<<<cdiv(NG * HD, 256), 256, 0, stream>>>(psum, pcnt, NG * HD);

  // MLP head: relu(pooled @ Wf1 + bf1) @ Wf2 + bf2
  k_gemm_wmma<HD, 128><<<cdiv(NG, 64), 128, 0, stream>>>(psum, Wf1, mlp1, NG);
  k_bias_relu<<<cdiv(NG * 128, 256), 256, 0, stream>>>(mlp1, bf1, NG * 128, 127);
  k_final    <<<cdiv(NG, 256), 256, 0, stream>>>(mlp1, Wf2, bf2, out, NG);
}
